// MultiheadAttention_15341623181846
// MI455X (gfx1250) — matrix operand order repaired and checked by reading, not compiled, not run on hardware
//
#include <hip/hip_runtime.h>
#include <stdint.h>

// Problem sizes (fixed by the reference)
#define NN 4096   // tokens
#define EE 1024   // embed dim
#define HH 16     // heads
#define DD 64     // head dim (DK == DV)

typedef __attribute__((ext_vector_type(8)))  float        v8f;
typedef __attribute__((ext_vector_type(8)))  __bf16       bf16x8;
typedef __attribute__((ext_vector_type(16))) __bf16       bf16x16;
typedef __attribute__((ext_vector_type(4)))  unsigned int u32x4;
typedef __attribute__((ext_vector_type(4)))  int          i32x4;

// ---------------------------------------------------------------------------
// gfx1250 async global->LDS copy (guarded: falls back to load+ds_store)
// ---------------------------------------------------------------------------
#if defined(__has_builtin)
#if __has_builtin(__builtin_amdgcn_global_load_async_to_lds_b128)
#define HAVE_ASYNC_COPY 1
#endif
#endif

static __device__ __forceinline__ void copy16_g2l(const unsigned short* g,
                                                  unsigned short* l) {
#if defined(HAVE_ASYNC_COPY)
  __builtin_amdgcn_global_load_async_to_lds_b128(
      (__attribute__((address_space(1))) i32x4*)(uintptr_t)g,
      (__attribute__((address_space(3))) i32x4*)(unsigned int)(uintptr_t)l,
      0, 0);
#else
  *reinterpret_cast<u32x4*>(l) = *reinterpret_cast<const u32x4*>(g);
#endif
}

static __device__ __forceinline__ void wait_async0() {
#if defined(HAVE_ASYNC_COPY)
#if __has_builtin(__builtin_amdgcn_s_wait_asynccnt)
  __builtin_amdgcn_s_wait_asynccnt(0);
#else
  asm volatile("s_wait_asynccnt 0x0" ::: "memory");
#endif
#endif
}

// ---------------------------------------------------------------------------
// helpers
// ---------------------------------------------------------------------------
static __device__ __forceinline__ unsigned short f2bf(float f) {
  unsigned int u = __builtin_bit_cast(unsigned int, f);
  unsigned int r = u + 0x7FFFu + ((u >> 16) & 1u);   // round-to-nearest-even
  return (unsigned short)(r >> 16);
}

static __device__ __forceinline__ v8f wmma_bf16(bf16x16 a, bf16x16 b, v8f c) {
  return __builtin_amdgcn_wmma_f32_16x16x32_bf16(false, a, false, b,
                                                 (short)0, c, false, false);
}

static __device__ __forceinline__ bf16x16 cat8(bf16x8 lo, bf16x8 hi) {
  bf16x16 a;
#pragma unroll
  for (int i = 0; i < 8; ++i) { a[i] = lo[i]; a[8 + i] = hi[i]; }
  return a;
}

// A fragment (16 x K=32): rows striped per lane, K split {0-7,16-23}/{8-15,24-31}
static __device__ __forceinline__ bf16x16
loadA(const unsigned short* base, int row0, int ld, int k0, int l16, int half) {
  const unsigned short* p = base + (size_t)(row0 + l16) * ld + k0 + half * 8;
  return cat8(*reinterpret_cast<const bf16x8*>(p),
              *reinterpret_cast<const bf16x8*>(p + 16));
}

// Same A fragment but from an LDS tile with leading dim ldc
static __device__ __forceinline__ bf16x16
loadA_lds(const unsigned short* base, int row, int ldc, int k0, int half) {
  const unsigned short* p = base + row * ldc + k0 + half * 8;
  return cat8(*reinterpret_cast<const bf16x8*>(p),
              *reinterpret_cast<const bf16x8*>(p + 16));
}

// B fragment (K=32 x 16 cols): lane = column, K split {0-7,16-23}/{8-15,24-31}.
// base is "transposed": row n = output column, cols = contraction dim.
static __device__ __forceinline__ bf16x16
loadB(const unsigned short* base, int n0, int ld, int k0, int l16, int half) {
  const unsigned short* p = base + (size_t)(n0 + l16) * ld + k0 + half * 8;
  return cat8(*reinterpret_cast<const bf16x8*>(p),
              *reinterpret_cast<const bf16x8*>(p + 16));
}

// ---------------------------------------------------------------------------
// 1) fp32 -> bf16 conversion kernels
// ---------------------------------------------------------------------------
__global__ void mha_cvt_copy(const float* __restrict__ src,
                             unsigned short* __restrict__ dst, int n) {
  int i = blockIdx.x * blockDim.x + threadIdx.x;
  if (i < n) dst[i] = f2bf(src[i]);
}

// dst[b][c][r] = src[b][r][c] * scale   (batched transpose + cast)
__global__ void mha_cvt_transpose(const float* __restrict__ src,
                                  unsigned short* __restrict__ dst,
                                  int B, int R, int C, float scale) {
  int i = blockIdx.x * blockDim.x + threadIdx.x;
  int total = B * R * C;
  if (i >= total) return;
  int b = i / (R * C);
  int rem = i - b * (R * C);
  int r = rem / C;
  int c = rem - r * C;
  dst[(size_t)b * R * C + (size_t)c * R + r] = f2bf(src[i] * scale);
}

// ---------------------------------------------------------------------------
// 2) QKV projection.  Block = 4 waves = 64 tokens; blockIdx.y picks Q/K/V.
//    x rows are staged through a double-buffered LDS tile (64 x 256 bf16 per
//    buffer) with async global->LDS copies; A-fragments then come from LDS.
//    Q,K stored [H][N][64]; V stored transposed [H][64][N].
// ---------------------------------------------------------------------------
#define XCH 256   // staged contraction chunk (elements)

__global__ __launch_bounds__(128)
void mha_qkv_proj(const unsigned short* __restrict__ xb,
                  const unsigned short* __restrict__ Wqt,
                  const unsigned short* __restrict__ Wkt,
                  const unsigned short* __restrict__ Wvt,
                  unsigned short* __restrict__ Qb,
                  unsigned short* __restrict__ Kb,
                  unsigned short* __restrict__ VTb) {
  __shared__ unsigned short smx[2][64 * XCH];   // 2 x 32 KB

  const int tid  = threadIdx.x;
  const int w    = tid >> 5;                    // wave 0..3 -> token sub-tile
  const int lane = tid & 31;
  const int half = lane >> 4, l16 = lane & 15;
  const int which = blockIdx.y;                 // 0=Q 1=K 2=V
  const int h  = blockIdx.x >> 6;               // 64 token-blocks per head
  const int tb = blockIdx.x & 63;               // 64-token block
  const int t0 = tb * 64;

  const unsigned short* Wt =
      ((which == 0) ? Wqt : (which == 1) ? Wkt : Wvt) + (size_t)h * DD * EE;

  // stage chunk: 64 rows x XCH elems, 16 b128 transfers per thread
  auto stage = [&](int buf, int ebase) {
    const unsigned short* gb = xb + (size_t)t0 * EE + ebase;
    unsigned short* lb = &smx[buf][0];
#pragma unroll
    for (int j = 0; j < 16; ++j) {
      int q   = tid + 128 * j;                  // 0..2047 chunks of 8 elems
      int row = q >> 5;
      int c8  = (q & 31) * 8;
      copy16_g2l(gb + (size_t)row * EE + c8, lb + row * XCH + c8);
    }
  };

  stage(0, 0);

  v8f c[4] = {};
  bf16x16 bcur[4];
#pragma unroll
  for (int t = 0; t < 4; ++t) bcur[t] = loadB(Wt, 16 * t, EE, 0, l16, half);

  wait_async0();
  __syncthreads();

  for (int s = 0; s < EE / XCH; ++s) {
    const int ebase = s * XCH;
    if (s + 1 < EE / XCH) stage((s + 1) & 1, ebase + XCH);   // async prefetch
    const unsigned short* lb = &smx[s & 1][0];

#pragma unroll
    for (int eo = 0; eo < XCH; eo += 32) {
      int e0 = ebase + eo;
      int en = (e0 + 32 < EE) ? e0 + 32 : e0;                // clamped prefetch
      bf16x16 bnx[4];
#pragma unroll
      for (int t = 0; t < 4; ++t) bnx[t] = loadB(Wt, 16 * t, EE, en, l16, half);

      bf16x16 a = loadA_lds(lb, 16 * w + l16, XCH, eo, half);
#pragma unroll
      for (int t = 0; t < 4; ++t) c[t] = wmma_bf16(a, bcur[t], c[t]);
#pragma unroll
      for (int t = 0; t < 4; ++t) bcur[t] = bnx[t];
    }

    if (s + 1 < EE / XCH) {
      wait_async0();
      __syncthreads();
    }
  }

  if (which < 2) {
    unsigned short* Dst = ((which == 0) ? Qb : Kb) + (size_t)h * NN * DD;
#pragma unroll
    for (int t = 0; t < 4; ++t)
#pragma unroll
      for (int i = 0; i < 8; ++i) {
        int row = t0 + 16 * w + i + 8 * half;               // token
        Dst[(size_t)row * DD + 16 * t + l16] = f2bf(c[t][i]);
      }
  } else {
    unsigned short* Dst = VTb + (size_t)h * DD * NN;        // [dv][token]
#pragma unroll
    for (int t = 0; t < 4; ++t)
#pragma unroll
      for (int i = 0; i < 8; ++i) {
        int row = t0 + 16 * w + i + 8 * half;
        Dst[(size_t)(16 * t + l16) * NN + row] = f2bf(c[t][i]);
      }
  }
}

// ---------------------------------------------------------------------------
// 3) Fused flash attention: one wave = 16 queries of one head vs all 4096 keys.
//    Computes S^T = K_tile x Q^T so exp(S) lands directly in the A-fragment
//    layout of the P@V WMMA.  K fragments are register double-buffered; V is loaded per block.
// ---------------------------------------------------------------------------
__global__ __launch_bounds__(32)
void mha_flash_attn(const unsigned short* __restrict__ Qb,
                    const unsigned short* __restrict__ Kb,
                    const unsigned short* __restrict__ VTb,
                    unsigned short* __restrict__ ATb) {
  const int lane = threadIdx.x & 31;
  const int half = lane >> 4, l16 = lane & 15;
  const int h  = blockIdx.x >> 8;
  const int qb = blockIdx.x & 255;

  const unsigned short* Qh = Qb  + (size_t)h * NN * DD;
  const unsigned short* Kh = Kb  + (size_t)h * NN * DD;
  const unsigned short* Vh = VTb + (size_t)h * DD * NN;

  // Q^T B-fragments, held across the key loop (1/8 scale folded into Wq)
  bf16x16 bq0 = loadB(Qh, qb * 16, DD, 0,  l16, half);
  bf16x16 bq1 = loadB(Qh, qb * 16, DD, 32, l16, half);

  // preload K fragments for first key block
  bf16x16 k0 = loadA(Kh, 0,  DD, 0,  l16, half);
  bf16x16 k1 = loadA(Kh, 0,  DD, 32, l16, half);
  bf16x16 k2 = loadA(Kh, 16, DD, 0,  l16, half);
  bf16x16 k3 = loadA(Kh, 16, DD, 32, l16, half);

  v8f o[4] = {};
  float m = -3.0e38f, lsum = 0.0f;

  for (int kb = 0; kb < NN; kb += 32) {
    const int kn = (kb + 32 < NN) ? kb + 32 : kb;           // clamped prefetch

    // issue V (current) and K (next) loads up front; they complete while the
    // S^T WMMAs and softmax VALU work run
    bf16x16 bv0 = loadB(Vh, 0,  NN, kb, l16, half);
    bf16x16 bv1 = loadB(Vh, 16, NN, kb, l16, half);
    bf16x16 bv2 = loadB(Vh, 32, NN, kb, l16, half);
    bf16x16 bv3 = loadB(Vh, 48, NN, kb, l16, half);
    bf16x16 n0 = loadA(Kh, kn,      DD, 0,  l16, half);
    bf16x16 n1 = loadA(Kh, kn,      DD, 32, l16, half);
    bf16x16 n2 = loadA(Kh, kn + 16, DD, 0,  l16, half);
    bf16x16 n3 = loadA(Kh, kn + 16, DD, 32, l16, half);

    v8f c0 = {}, c1 = {};
    c0 = wmma_bf16(k0, bq0, c0); c0 = wmma_bf16(k1, bq1, c0);
    c1 = wmma_bf16(k2, bq0, c1); c1 = wmma_bf16(k3, bq1, c1);

    // online softmax; per-lane row (query = l16), partner lane = lane^16
    float bm = c0[0];
#pragma unroll
    for (int i = 0; i < 8; ++i) { bm = fmaxf(bm, c0[i]); bm = fmaxf(bm, c1[i]); }
    bm = fmaxf(bm, __shfl_xor(bm, 16, 32));
    float mn    = fmaxf(m, bm);
    float alpha = __expf(m - mn);

    float rs = 0.0f;
    bf16x16 pa;                       // P already in A-fragment layout
#pragma unroll
    for (int i = 0; i < 8; ++i) {
      float p0 = __expf(c0[i] - mn); rs += p0;
      float p1 = __expf(c1[i] - mn); rs += p1;
      pa[i]     = __builtin_bit_cast(__bf16, f2bf(p0));
      pa[8 + i] = __builtin_bit_cast(__bf16, f2bf(p1));
    }
    rs += __shfl_xor(rs, 16, 32);
    lsum = lsum * alpha + rs;
    m = mn;

    // rescale O by alpha of its row (row = reg + 8*half in C layout)
#pragma unroll
    for (int i = 0; i < 8; ++i) {
      float af = __shfl(alpha, i + 8 * half, 32);
      o[0][i] *= af; o[1][i] *= af; o[2][i] *= af; o[3][i] *= af;
    }

    o[0] = wmma_bf16(pa, bv0, o[0]);
    o[1] = wmma_bf16(pa, bv1, o[1]);
    o[2] = wmma_bf16(pa, bv2, o[2]);
    o[3] = wmma_bf16(pa, bv3, o[3]);

    k0 = n0; k1 = n1; k2 = n2; k3 = n3;
  }

  // normalize and write concatenated head output [N][H*DV] as bf16
  float inv = 1.0f / lsum;
#pragma unroll
  for (int i = 0; i < 8; ++i) {
    float fi = __shfl(inv, i + 8 * half, 32);
    int row  = qb * 16 + i + 8 * half;
#pragma unroll
    for (int t = 0; t < 4; ++t)
      ATb[(size_t)row * (HH * DD) + h * DD + 16 * t + l16] = f2bf(o[t][i] * fi);
  }
}

// ---------------------------------------------------------------------------
// 4) Output projection: out[N][1024] (fp32) = AT[N][1024] @ Wo[1024][1024]
//    Register double-buffered A and B fragments.
// ---------------------------------------------------------------------------
__global__ __launch_bounds__(32)
void mha_out_proj(const unsigned short* __restrict__ ATb,
                  const unsigned short* __restrict__ Wot,   // [outcol][hdv]
                  float* __restrict__ out) {
  const int lane = threadIdx.x & 31;
  const int half = lane >> 4, l16 = lane & 15;
  const int tb = blockIdx.x;     // 256 token tiles
  const int cb = blockIdx.y;     // 16 column groups of 64

  v8f c[4] = {};
  bf16x16 acur = loadA(ATb, tb * 16, EE, 0, l16, half);
  bf16x16 bcur[4];
#pragma unroll
  for (int t = 0; t < 4; ++t)
    bcur[t] = loadB(Wot, cb * 64 + 16 * t, EE, 0, l16, half);

  for (int e0 = 0; e0 < EE; e0 += 32) {
    const int en = (e0 + 32 < EE) ? e0 + 32 : e0;           // clamped prefetch
    bf16x16 anx = loadA(ATb, tb * 16, EE, en, l16, half);
    bf16x16 bnx[4];
#pragma unroll
    for (int t = 0; t < 4; ++t)
      bnx[t] = loadB(Wot, cb * 64 + 16 * t, EE, en, l16, half);

#pragma unroll
    for (int t = 0; t < 4; ++t) c[t] = wmma_bf16(acur, bcur[t], c[t]);

    acur = anx;
#pragma unroll
    for (int t = 0; t < 4; ++t) bcur[t] = bnx[t];
  }

#pragma unroll
  for (int t = 0; t < 4; ++t)
#pragma unroll
    for (int i = 0; i < 8; ++i) {
      int row = tb * 16 + i + 8 * half;
      out[(size_t)row * EE + cb * 64 + 16 * t + l16] = c[t][i];
    }
}

// ---------------------------------------------------------------------------
// host-side launcher
// ---------------------------------------------------------------------------
extern "C" void kernel_launch(void* const* d_in, const int* in_sizes, int n_in,
                              void* d_out, int out_size, void* d_ws, size_t ws_size,
                              hipStream_t stream) {
  (void)in_sizes; (void)n_in; (void)out_size; (void)ws_size;
  const float* x  = (const float*)d_in[0];   // [N][E]
  const float* Wq = (const float*)d_in[1];   // [H][E][DK]
  const float* Wk = (const float*)d_in[2];   // [H][E][DK]
  const float* Wv = (const float*)d_in[3];   // [H][E][DV]
  const float* Wo = (const float*)d_in[4];   // [H*DV][E]
  float* out = (float*)d_out;                // [N][E] fp32

  char* ws = (char*)d_ws;
  size_t off = 0;
  auto carve = [&](size_t bytes) -> unsigned short* {
    unsigned short* p = (unsigned short*)(ws + off);
    off += (bytes + 255) & ~(size_t)255;
    return p;
  };
  unsigned short* xb  = carve((size_t)NN * EE * 2);        //  8 MB
  unsigned short* Wqt = carve((size_t)HH * DD * EE * 2);   //  2 MB (scaled 1/8)
  unsigned short* Wkt = carve((size_t)HH * DD * EE * 2);   //  2 MB
  unsigned short* Wvt = carve((size_t)HH * DD * EE * 2);   //  2 MB
  unsigned short* Wot = carve((size_t)EE * EE * 2);        //  2 MB
  unsigned short* Qb  = carve((size_t)HH * NN * DD * 2);   //  8 MB
  unsigned short* Kb  = carve((size_t)HH * NN * DD * 2);   //  8 MB
  unsigned short* VTb = carve((size_t)HH * DD * NN * 2);   //  8 MB
  unsigned short* ATb = carve((size_t)NN * (HH * DD) * 2); //  8 MB

  int n;
  n = NN * EE;
  mha_cvt_copy<<<(n + 255) / 256, 256, 0, stream>>>(x, xb, n);
  n = HH * EE * DD;
  mha_cvt_transpose<<<(n + 255) / 256, 256, 0, stream>>>(Wq, Wqt, HH, EE, DD, 0.125f);
  mha_cvt_transpose<<<(n + 255) / 256, 256, 0, stream>>>(Wk, Wkt, HH, EE, DD, 1.0f);
  mha_cvt_transpose<<<(n + 255) / 256, 256, 0, stream>>>(Wv, Wvt, HH, EE, DD, 1.0f);
  n = EE * EE;
  mha_cvt_transpose<<<(n + 255) / 256, 256, 0, stream>>>(Wo, Wot, 1, EE, EE, 1.0f);

  dim3 gq(HH * (NN / 64), 3);
  mha_qkv_proj<<<gq, 128, 0, stream>>>(xb, Wqt, Wkt, Wvt, Qb, Kb, VTb);

  mha_flash_attn<<<HH * (NN / 16), 32, 0, stream>>>(Qb, Kb, VTb, ATb);

  dim3 go(NN / 16, EE / 64);
  mha_out_proj<<<go, 32, 0, stream>>>(ATb, Wot, out);
}
